// SampleAttention_30150670418306
// MI455X (gfx1250) — matrix operand order repaired and checked by reading, not compiled, not run on hardware
//
#include <hip/hip_runtime.h>

typedef __attribute__((ext_vector_type(16))) __bf16 v16bf;
typedef __attribute__((ext_vector_type(8)))  __bf16 v8bf;
typedef __attribute__((ext_vector_type(8)))  float  v8f;

#define NUM_HEADS 12
#define N_POINTS  4
#define BATCH     2
#define HH        64
#define WW        64
#define NPIX      (HH*WW)        // 4096
#define CDIM      768
#define HD        64             // head dim
#define MROWS     (BATCH*NPIX)   // 8192
#define QKVN      (3*CDIM)       // 2304

// ---------------------------------------------------------------------------
// fp32 -> bf16 elementwise convert
// ---------------------------------------------------------------------------
__global__ void k_f32_to_bf16(const float* __restrict__ in, __bf16* __restrict__ out, int n) {
  int i = blockIdx.x * blockDim.x + threadIdx.x;
  if (i < n) out[i] = (__bf16)in[i];
}

// in: [K][N] fp32  ->  out: [N][K] bf16  (small weight matrices only)
__global__ void k_transpose_f32_to_bf16(const float* __restrict__ in, __bf16* __restrict__ out,
                                        int K, int N) {
  int i = blockIdx.x * blockDim.x + threadIdx.x;
  if (i < K * N) {
    int k = i / N, n = i - k * N;
    out[(size_t)n * K + k] = (__bf16)in[i];
  }
}

// ---------------------------------------------------------------------------
// CDNA5 async global->LDS copy (16B per lane), tracked by ASYNCcnt.
// dst LDS offset = low 32 bits of the generic address (LDS aperture rule).
// ---------------------------------------------------------------------------
__device__ __forceinline__ void async_copy16(const __bf16* g, __bf16* l) {
  unsigned lds = (unsigned)(unsigned long long)l;
  unsigned long long ga = (unsigned long long)g;
  asm volatile("global_load_async_to_lds_b128 %0, %1, off"
               :: "v"(lds), "v"(ga) : "memory");
}
__device__ __forceinline__ void wait_async_le(void) {  // tile k landed, k+1 in flight
  asm volatile("s_wait_asynccnt 0x8" ::: "memory");
}
__device__ __forceinline__ void wait_async_0(void) {
  asm volatile("s_wait_asynccnt 0x0" ::: "memory");
}

// ---------------------------------------------------------------------------
// bf16 WMMA GEMM:  C[M][Nn] = A[M][K] @ Bt[Nn][K]^T  (+ bias)
// Block tile 128x128, K-tile 64 (2 WMMA K-steps), 256 threads = 8 waves.
// Double-buffered LDS filled by global_load_async_to_lds_b128 (ASYNCcnt).
// Wave (wm,wn) in 4x2 grid computes a 32x64 region = 2x4 WMMA 16x16 tiles.
// Assumes M % 128 == 0 and K % 64 == 0 (true for all uses: K == 768).
// B rows beyond Nn are DMA'd from adjacent (valid) workspace and only feed
// accumulator columns that are never stored (store is guarded by col < Nn).
// ---------------------------------------------------------------------------
#define TM 128
#define TN 128
#define TK 64
#define LDST 72   // bf16 elements per LDS row (144B = 36 banks -> conflict-free)

__device__ __forceinline__ void issue_tile_copies(
    const __bf16* __restrict__ A, int lda,
    const __bf16* __restrict__ Bt, int ldb,
    __bf16* lAbuf, __bf16* lBbuf,
    int m0, int n0, int k0, int t) {
#pragma unroll
  for (int i = 0; i < 4; ++i) {
    int c = t + i * 256;          // 1024 chunks of 16B per matrix tile
    int r = c >> 3, s = c & 7;    // row 0..127, 8 chunks per row
    async_copy16(A + (size_t)(m0 + r) * lda + (k0 + s * 8), lAbuf + r * LDST + s * 8);
    async_copy16(Bt + (size_t)(n0 + r) * ldb + (k0 + s * 8), lBbuf + r * LDST + s * 8);
  }
}

__global__ __launch_bounds__(256)
void k_wmma_gemm(const __bf16* __restrict__ A, int lda,
                 const __bf16* __restrict__ Bt, int ldb,
                 const float* __restrict__ bias,
                 float* __restrict__ Cf, __bf16* __restrict__ Cb, int ldc,
                 int Nn, int K) {
  __shared__ __bf16 lA[2][TM * LDST];
  __shared__ __bf16 lB[2][TN * LDST];

  const int t    = threadIdx.x;
  const int lane = t & 31;
  const int w    = t >> 5;
  const int wm   = w & 3;        // 0..3  (row group of 32)
  const int wn   = w >> 2;       // 0..1  (col group of 64)
  const int m0   = blockIdx.x * TM;
  const int n0   = blockIdx.y * TN;

  const int mrow = lane & 15;            // row/col within 16x16 tile
  const int kbA  = (lane >> 4) * 8;      // A-fragment K base (ISA layout)
  const int kbB  = (lane >> 4) * 8;      // B-fragment K base (ISA layout)

  v8f acc[2][4];
#pragma unroll
  for (int mt = 0; mt < 2; ++mt)
#pragma unroll
    for (int nt = 0; nt < 4; ++nt)
#pragma unroll
      for (int r = 0; r < 8; ++r) acc[mt][nt][r] = 0.0f;

  const int nk = K / TK;
  issue_tile_copies(A, lda, Bt, ldb, lA[0], lB[0], m0, n0, 0, t);

  for (int kk = 0; kk < nk; ++kk) {
    __bf16* curA = lA[kk & 1];
    __bf16* curB = lB[kk & 1];
    if (kk + 1 < nk) {
      issue_tile_copies(A, lda, Bt, ldb, lA[(kk + 1) & 1], lB[(kk + 1) & 1],
                        m0, n0, (kk + 1) * TK, t);
      wait_async_le();   // our 8 copies for tile kk have landed
    } else {
      wait_async_0();
    }
    __syncthreads();     // all waves' copies for tile kk landed

#pragma unroll
    for (int ks = 0; ks < TK; ks += 32) {
      v16bf af[2], bfr[4];
#pragma unroll
      for (int mt = 0; mt < 2; ++mt) {
        const __bf16* p = &curA[(wm * 32 + mt * 16 + mrow) * LDST + ks];
        v8bf a0 = *(const v8bf*)(p + kbA);
        v8bf a1 = *(const v8bf*)(p + kbA + 16);
        af[mt] = __builtin_shufflevector(a0, a1, 0,1,2,3,4,5,6,7,8,9,10,11,12,13,14,15);
      }
#pragma unroll
      for (int nt = 0; nt < 4; ++nt) {
        const __bf16* p = &curB[(wn * 64 + nt * 16 + mrow) * LDST + ks];
        v8bf b0 = *(const v8bf*)(p + kbB);
        v8bf b1 = *(const v8bf*)(p + kbB + 16);
        bfr[nt] = __builtin_shufflevector(b0, b1, 0,1,2,3,4,5,6,7,8,9,10,11,12,13,14,15);
      }
#pragma unroll
      for (int mt = 0; mt < 2; ++mt)
#pragma unroll
        for (int nt = 0; nt < 4; ++nt)
          acc[mt][nt] = __builtin_amdgcn_wmma_f32_16x16x32_bf16(
              false, af[mt], false, bfr[nt], (short)0, acc[mt][nt], false, false);
    }
    __syncthreads();     // reads of tile kk done before buffer reuse at kk+2
  }

  // Store: C/D layout: VGPR r, lanes0-15 -> M=r, lanes16-31 -> M=r+8; N=lane%16
#pragma unroll
  for (int mt = 0; mt < 2; ++mt) {
#pragma unroll
    for (int nt = 0; nt < 4; ++nt) {
      int col = n0 + wn * 64 + nt * 16 + (lane & 15);
      if (col < Nn) {
        float bv = bias ? bias[col] : 0.0f;
#pragma unroll
        for (int r = 0; r < 8; ++r) {
          int row = m0 + wm * 32 + mt * 16 + r + ((lane >> 4) << 3);
          float v = acc[mt][nt][r] + bv;
          if (Cf) Cf[(size_t)row * ldc + col] = v;
          if (Cb) Cb[(size_t)row * ldc + col] = (__bf16)v;
        }
      }
    }
  }
}

// ---------------------------------------------------------------------------
// Grid-sample + 4-point attention. One wave32 per (b, head, n).
// qkvb: bf16 [8192][2304] (q|k|v), off: f32 [8192][96], refp: f32 [8192][2]
// outb: bf16 [8192][768]
// Lane l owns channels d=l and d=l+32 of the 64-dim head.
// ---------------------------------------------------------------------------
__global__ __launch_bounds__(256)
void k_sample_attn(const __bf16* __restrict__ qkvb,
                   const float* __restrict__ off,
                   const float* __restrict__ refp,
                   __bf16* __restrict__ outb) {
  const int lane = threadIdx.x & 31;
  const int wid  = blockIdx.x * (blockDim.x >> 5) + (threadIdx.x >> 5);
  if (wid >= BATCH * NUM_HEADS * NPIX) return;

  const int n    = wid & (NPIX - 1);
  const int bh   = wid >> 12;           // NPIX == 4096 == 2^12
  const int head = bh % NUM_HEADS;
  const int b    = bh / NUM_HEADS;
  const size_t row = (size_t)(b * NPIX + n);

  const __bf16* qrow = qkvb + row * QKVN + head * HD;
  const float q0 = (float)qrow[lane];
  const float q1 = (float)qrow[lane + 32];

  const float rx = refp[row * 2 + 0];
  const float ry = refp[row * 2 + 1];
  const float* offp = off + row * 96 + (head * N_POINTS) * 2;

  float dotv[N_POINTS];
  float svv[N_POINTS][2];

#pragma unroll
  for (int p = 0; p < N_POINTS; ++p) {
    // loc in [-1,1], then unnormalized pixel coords (align_corners=False)
    float gx = 2.0f * (rx + offp[2 * p + 0] * (1.0f / (float)WW)) - 1.0f;
    float gy = 2.0f * (ry + offp[2 * p + 1] * (1.0f / (float)HH)) - 1.0f;
    float ix = (gx + 1.0f) * ((float)WW * 0.5f) - 0.5f;
    float iy = (gy + 1.0f) * ((float)HH * 0.5f) - 0.5f;
    float x0f = floorf(ix), y0f = floorf(iy);
    float wx = ix - x0f, wy = iy - y0f;

    float sk0 = 0.f, sk1 = 0.f, sv0 = 0.f, sv1 = 0.f;
#pragma unroll
    for (int cy = 0; cy < 2; ++cy) {
#pragma unroll
      for (int cx = 0; cx < 2; ++cx) {
        float xc = x0f + (float)cx, yc = y0f + (float)cy;
        float wgt = (cx ? wx : 1.0f - wx) * (cy ? wy : 1.0f - wy);
        // lane-uniform condition: zero padding outside the map
        if (xc >= 0.0f && xc <= (float)(WW - 1) && yc >= 0.0f && yc <= (float)(HH - 1)) {
          int xi = (int)xc, yi = (int)yc;
          size_t base = ((size_t)(b * NPIX + yi * WW + xi)) * QKVN + CDIM + head * HD;
          sk0 += wgt * (float)qkvb[base + lane];
          sk1 += wgt * (float)qkvb[base + lane + 32];
          sv0 += wgt * (float)qkvb[base + CDIM + lane];
          sv1 += wgt * (float)qkvb[base + CDIM + lane + 32];
        }
      }
    }
    svv[p][0] = sv0; svv[p][1] = sv1;

    float d = q0 * sk0 + q1 * sk1;       // partial dot over this lane's 2 dims
#pragma unroll
    for (int m = 16; m > 0; m >>= 1) d += __shfl_xor(d, m, 32);
    dotv[p] = d * 0.125f;                // hd^-0.5 = 64^-0.5
  }

  // softmax over the 4 points (identical in all lanes after full butterfly)
  float mx = fmaxf(fmaxf(dotv[0], dotv[1]), fmaxf(dotv[2], dotv[3]));
  float e[N_POINTS], s = 0.f;
#pragma unroll
  for (int p = 0; p < N_POINTS; ++p) { e[p] = expf(dotv[p] - mx); s += e[p]; }
  float inv = 1.0f / s;

  float o0 = 0.f, o1 = 0.f;
#pragma unroll
  for (int p = 0; p < N_POINTS; ++p) {
    float a = e[p] * inv;
    o0 += a * svv[p][0];
    o1 += a * svv[p][1];
  }
  __bf16* orow = outb + row * CDIM + head * HD;
  orow[lane]      = (__bf16)o0;
  orow[lane + 32] = (__bf16)o1;
}

// ---------------------------------------------------------------------------
// launch
// ---------------------------------------------------------------------------
extern "C" void kernel_launch(void* const* d_in, const int* in_sizes, int n_in,
                              void* d_out, int out_size, void* d_ws, size_t ws_size,
                              hipStream_t stream) {
  const float* x     = (const float*)d_in[0];
  const float* refp  = (const float*)d_in[1];
  const float* Wqkv  = (const float*)d_in[2];
  const float* Woff  = (const float*)d_in[3];
  const float* boff  = (const float*)d_in[4];
  const float* Wproj = (const float*)d_in[5];
  const float* bproj = (const float*)d_in[6];
  // d_in[7]=H, d_in[8]=W (fixed 64x64 per setup_inputs)

  char* ws = (char*)d_ws;
  __bf16* xb     = (__bf16*)(ws);                 // 12,582,912 B
  __bf16* qkvb   = (__bf16*)(ws + 12582912);      // 37,748,736 B
  __bf16* Wqkvt  = (__bf16*)(ws + 50331648);      //  3,538,944 B
  __bf16* Wofft  = (__bf16*)(ws + 53870592);      //    147,456 B
  __bf16* Wprojt = (__bf16*)(ws + 54018048);      //  1,179,648 B
  float*  offbuf = (float*) (ws + 55197696);      //  3,145,728 B
  __bf16* outb   = (__bf16*)(ws + 58343424);      // 12,582,912 B  (total ~71 MB)

  // bf16 conversions / weight transposes (tiny vs. GEMM cost)
  k_f32_to_bf16<<<(MROWS * CDIM + 255) / 256, 256, 0, stream>>>(x, xb, MROWS * CDIM);
  k_transpose_f32_to_bf16<<<(CDIM * QKVN + 255) / 256, 256, 0, stream>>>(Wqkv, Wqkvt, CDIM, QKVN);
  k_transpose_f32_to_bf16<<<(CDIM * 96 + 255) / 256, 256, 0, stream>>>(Woff, Wofft, CDIM, 96);
  k_transpose_f32_to_bf16<<<(CDIM * CDIM + 255) / 256, 256, 0, stream>>>(Wproj, Wprojt, CDIM, CDIM);

  // qkv = x @ Wqkv  -> bf16 [8192][2304]
  k_wmma_gemm<<<dim3(MROWS / 128, QKVN / 128), 256, 0, stream>>>(
      xb, CDIM, Wqkvt, CDIM, nullptr, nullptr, qkvb, QKVN, QKVN, CDIM);

  // off = q @ Woff + boff -> f32 [8192][96]  (A = strided q view of qkvb)
  k_wmma_gemm<<<dim3(MROWS / 128, 1), 256, 0, stream>>>(
      qkvb, QKVN, Wofft, CDIM, boff, offbuf, nullptr, 96, 96, CDIM);

  // grid-sample + attention -> bf16 [8192][768]
  k_sample_attn<<<(BATCH * NUM_HEADS * NPIX) / 8, 256, 0, stream>>>(qkvb, offbuf, refp, outb);

  // out = attn_out @ Wproj + bproj -> f32 d_out
  k_wmma_gemm<<<dim3(MROWS / 128, CDIM / 128), 256, 0, stream>>>(
      outb, CDIM, Wprojt, CDIM, bproj, (float*)d_out, nullptr, CDIM, CDIM, CDIM);
}
